// GPRGNN_4209067950740
// MI455X (gfx1250) — compile-verified, matrix operand order checked by reading, not run on hardware
//
#include <hip/hip_runtime.h>

#define F_DIM 512
#define H_DIM 256
#define C_DIM 16
#define K_HOPS 10

typedef __attribute__((ext_vector_type(16))) __bf16 v16bf;
typedef __attribute__((ext_vector_type(8)))  float  v8f;

// ---------- helpers ----------
__device__ __forceinline__ unsigned int bf_bits(float f) {
  unsigned int u = __builtin_bit_cast(unsigned int, f);
  return (u + 0x7FFFu + ((u >> 16) & 1u)) >> 16;   // round-to-nearest-even bf16
}
__device__ __forceinline__ unsigned int pack_bf2(float lo, float hi) {
  return bf_bits(lo) | (bf_bits(hi) << 16);
}
__device__ __forceinline__ void atomicAddF(float* p, float v) {
  unsafeAtomicAdd(p, v);   // native global_atomic_add_f32 (no CAS loop)
}

// ---------- W1 [F,H] f32 -> W1^T [H,F] bf16 ----------
__global__ __launch_bounds__(256) void k_w1_transpose(const float* __restrict__ w1,
                                                      __bf16* __restrict__ w1t) {
  int idx = blockIdx.x * 256 + threadIdx.x;
  if (idx >= F_DIM * H_DIM) return;
  int f = idx >> 8;        // / H_DIM
  int h = idx & (H_DIM - 1);
  unsigned short b = (unsigned short)bf_bits(w1[idx]);
  w1t[(size_t)h * F_DIM + f] = __builtin_bit_cast(__bf16, b);
}

// ---------- GEMM1: h1 = relu(x @ W1 + b1), bf16 WMMA, f32 accum ----------
// block = 256 threads = 8 waves; block tile 64 rows x 128 cols;
// wave (wm=wid&3, wn=wid>>2) computes 16x64 via 4 accumulators.
__global__ __launch_bounds__(256) void k_mlp1_wmma(const float* __restrict__ x,
                                                   const __bf16* __restrict__ w1t,
                                                   const float* __restrict__ b1,
                                                   float* __restrict__ h1, int Nn) {
  const int tid  = threadIdx.x;
  const int lane = tid & 31;
  const int l15  = lane & 15;
  const int kb   = (lane >> 4) << 3;      // 0 or 8 (K sub-offset; also D row offset)
  const int wid  = tid >> 5;
  const int wm   = wid & 3;
  const int wn   = wid >> 2;
  const int bm   = blockIdx.x * 64;
  const int bn   = blockIdx.y * 128;
  const int col0 = bn + wn * 64;

  int row  = bm + wm * 16 + l15;
  int rowc = row < Nn ? row : (Nn - 1);   // clamp for tail block (stores are guarded)
  const float* xr = x + (size_t)rowc * F_DIM;

  v8f acc[4];
#pragma unroll
  for (int j = 0; j < 4; ++j) acc[j] = (v8f){0.f,0.f,0.f,0.f,0.f,0.f,0.f,0.f};

  for (int kk = 0; kk < F_DIM; kk += 32) {
    // A fragment: two contiguous 8-float runs, converted to bf16 in-register
    union { v16bf v; unsigned int u[8]; } fa;
    const float4* pa = (const float4*)(xr + kk + kb);
    float4 a0 = pa[0];
    float4 a1 = pa[1];
    const float4* pb = (const float4*)(xr + kk + kb + 16);
    float4 a2 = pb[0];
    float4 a3 = pb[1];
    fa.u[0] = pack_bf2(a0.x, a0.y); fa.u[1] = pack_bf2(a0.z, a0.w);
    fa.u[2] = pack_bf2(a1.x, a1.y); fa.u[3] = pack_bf2(a1.z, a1.w);
    fa.u[4] = pack_bf2(a2.x, a2.y); fa.u[5] = pack_bf2(a2.z, a2.w);
    fa.u[6] = pack_bf2(a3.x, a3.y); fa.u[7] = pack_bf2(a3.z, a3.w);

#pragma unroll
    for (int j = 0; j < 4; ++j) {
      // B fragment: column n = lane&15 is a row of W1^T -> two b128 loads
      const __bf16* wr = w1t + (size_t)(col0 + j * 16 + l15) * F_DIM + kk + kb;
      union { v16bf v; uint4 q[2]; } fb;
      fb.q[0] = *(const uint4*)(wr);
      fb.q[1] = *(const uint4*)(wr + 16);
      acc[j] = __builtin_amdgcn_wmma_f32_16x16x32_bf16(
          false, fa.v, false, fb.v, (short)0, acc[j], false, false);
    }
  }

  // epilogue: D layout -> row = base + kb + r, col = base + (lane&15)
#pragma unroll
  for (int j = 0; j < 4; ++j) {
    int ocol = col0 + j * 16 + l15;
    float bias = b1[ocol];
#pragma unroll
    for (int r = 0; r < 8; ++r) {
      int orow = bm + wm * 16 + kb + r;
      if (orow < Nn) {
        float v = acc[j][r] + bias;
        h1[(size_t)orow * H_DIM + ocol] = v > 0.f ? v : 0.f;
      }
    }
  }
}

// ---------- GEMM2: h = h1 @ W2 + b2 ; hcur = h ; out = temp[0]*h ----------
__global__ __launch_bounds__(256) void k_mlp2(const float* __restrict__ h1,
                                              const float* __restrict__ w2,
                                              const float* __restrict__ b2,
                                              const float* __restrict__ temp,
                                              float* __restrict__ hcur,
                                              float* __restrict__ out, int Nn) {
  __shared__ float w2s[H_DIM * C_DIM];   // 16 KB
  __shared__ float hs[16 * H_DIM];       // 16 KB
  const int tid  = threadIdx.x;
  const int row0 = blockIdx.x * 16;
#pragma unroll
  for (int j = 0; j < 16; ++j) w2s[tid + j * 256] = w2[tid + j * 256];
  const float* hb = h1 + (size_t)row0 * H_DIM;
#pragma unroll
  for (int j = 0; j < 16; ++j)
    if (row0 + j < Nn) hs[tid + j * 256] = hb[tid + j * 256];
  __syncthreads();

  const int r = tid >> 4;
  const int c = tid & 15;
  if (row0 + r >= Nn) return;
  float acc = b2[c];
  const float* hr = hs + r * H_DIM;
#pragma unroll 8
  for (int k = 0; k < H_DIM; ++k) acc = fmaf(hr[k], w2s[k * C_DIM + c], acc);
  size_t oidx = (size_t)(row0 + r) * C_DIM + c;
  hcur[oidx] = acc;
  out[oidx]  = temp[0] * acc;
}

// ---------- degree / dinv ----------
__global__ __launch_bounds__(256) void k_set1(float* __restrict__ d, int n) {
  int i = blockIdx.x * 256 + threadIdx.x;
  if (i < n) d[i] = 1.0f;                 // self-loop contribution
}
__global__ __launch_bounds__(256) void k_deg_edges(const int* __restrict__ ei,
                                                   float* __restrict__ deg, int Ee) {
  int e = blockIdx.x * 256 + threadIdx.x;
  if (e >= Ee) return;
  atomicAddF(deg + ei[Ee + e], 1.0f);     // col = target
}
__global__ __launch_bounds__(256) void k_rsqrt(float* __restrict__ d, int n) {
  int i = blockIdx.x * 256 + threadIdx.x;
  if (i < n) d[i] = rsqrtf(d[i]);         // deg >= 1 always
}

// ---------- propagation: one lane per (edge, channel) ----------
__global__ __launch_bounds__(256) void k_edge_prop(const int* __restrict__ ei,
                                                   const float* __restrict__ dinv,
                                                   const float* __restrict__ hin,
                                                   float* __restrict__ hout, int Ee) {
  int gid = blockIdx.x * 256 + threadIdx.x;
  int e = gid >> 4;
  int c = gid & 15;
  if (e >= Ee) return;
  int s = ei[e];            // source (row)
  int t = ei[Ee + e];       // target (col)
  float coef = dinv[s] * dinv[t];
  atomicAddF(hout + (size_t)t * C_DIM + c, coef * hin[(size_t)s * C_DIM + c]);
}

// self-loop term + GPR accumulation fused: hnext += dinv^2*hcur ; out += temp[k+1]*hnext
__global__ __launch_bounds__(256) void k_self_accum(const float* __restrict__ hin,
                                                    float* __restrict__ hnext,
                                                    const float* __restrict__ dinv,
                                                    const float* __restrict__ temp,
                                                    float* __restrict__ out,
                                                    int hop, int Nn) {
  int idx = blockIdx.x * 256 + threadIdx.x;
  if (idx >= Nn * C_DIM) return;
  int i = idx >> 4;
  float d = dinv[i];
  float v = hnext[idx] + d * d * hin[idx];
  hnext[idx] = v;
  out[idx] += temp[hop + 1] * v;
}

// ---------- log_softmax in place on [N,16] ----------
__global__ __launch_bounds__(256) void k_logsoftmax(float* __restrict__ out, int Nn) {
  int i = blockIdx.x * 256 + threadIdx.x;
  if (i >= Nn) return;
  float* p = out + (size_t)i * C_DIM;
  float v[16];
  float4* q = (float4*)p;
#pragma unroll
  for (int j = 0; j < 4; ++j) {
    float4 t = q[j];
    v[4*j] = t.x; v[4*j+1] = t.y; v[4*j+2] = t.z; v[4*j+3] = t.w;
  }
  float m = v[0];
#pragma unroll
  for (int j = 1; j < 16; ++j) m = fmaxf(m, v[j]);
  float s = 0.f;
#pragma unroll
  for (int j = 0; j < 16; ++j) s += __expf(v[j] - m);
  float ls = __logf(s) + m;
#pragma unroll
  for (int j = 0; j < 4; ++j) {
    float4 t;
    t.x = v[4*j] - ls; t.y = v[4*j+1] - ls; t.z = v[4*j+2] - ls; t.w = v[4*j+3] - ls;
    q[j] = t;
  }
}

// ---------- host orchestration ----------
extern "C" void kernel_launch(void* const* d_in, const int* in_sizes, int n_in,
                              void* d_out, int out_size, void* d_ws, size_t ws_size,
                              hipStream_t stream) {
  const float* x    = (const float*)d_in[0];
  const int*   ei   = (const int*)d_in[1];
  const float* w1   = (const float*)d_in[2];
  const float* b1   = (const float*)d_in[3];
  const float* w2   = (const float*)d_in[4];
  const float* b2   = (const float*)d_in[5];
  const float* temp = (const float*)d_in[6];
  float* out = (float*)d_out;

  const int Nn = in_sizes[0] / F_DIM;
  const int Ee = in_sizes[1] / 2;

  char* ws = (char*)d_ws;
  size_t off = 0;
  __bf16* w1t = (__bf16*)(ws + off); off += (size_t)F_DIM * H_DIM * sizeof(__bf16);
  off = (off + 255) & ~(size_t)255;
  float* h1 = (float*)(ws + off); off += (size_t)Nn * H_DIM * sizeof(float);
  off = (off + 255) & ~(size_t)255;
  float* ha = (float*)(ws + off); off += (size_t)Nn * C_DIM * sizeof(float);
  off = (off + 255) & ~(size_t)255;
  float* hbuf = (float*)(ws + off); off += (size_t)Nn * C_DIM * sizeof(float);
  off = (off + 255) & ~(size_t)255;
  float* dinv = (float*)(ws + off);

  // MLP
  k_w1_transpose<<<(F_DIM * H_DIM + 255) / 256, 256, 0, stream>>>(w1, w1t);
  dim3 g1((Nn + 63) / 64, H_DIM / 128);
  k_mlp1_wmma<<<g1, 256, 0, stream>>>(x, w1t, b1, h1, Nn);
  k_mlp2<<<(Nn + 15) / 16, 256, 0, stream>>>(h1, w2, b2, temp, ha, out, Nn);

  // gcn_norm: deg = 1 + indegree ; dinv = deg^-1/2
  k_set1<<<(Nn + 255) / 256, 256, 0, stream>>>(dinv, Nn);
  k_deg_edges<<<(Ee + 255) / 256, 256, 0, stream>>>(ei, dinv, Ee);
  k_rsqrt<<<(Nn + 255) / 256, 256, 0, stream>>>(dinv, Nn);

  // K-hop GPR propagation
  float* hcur = ha;
  float* hnext = hbuf;
  for (int hop = 0; hop < K_HOPS; ++hop) {
    hipMemsetAsync(hnext, 0, (size_t)Nn * C_DIM * sizeof(float), stream);
    long long tot = (long long)Ee * C_DIM;
    k_edge_prop<<<(unsigned)((tot + 255) / 256), 256, 0, stream>>>(ei, dinv, hcur, hnext, Ee);
    k_self_accum<<<(Nn * C_DIM + 255) / 256, 256, 0, stream>>>(hcur, hnext, dinv, temp, out,
                                                               hop, Nn);
    float* t = hcur; hcur = hnext; hnext = t;
  }

  k_logsoftmax<<<(Nn + 255) / 256, 256, 0, stream>>>(out, Nn);
}
